// HGraphConv_15187004358920
// MI455X (gfx1250) — matrix operand order repaired and checked by reading, not compiled, not run on hardware
//
#include <hip/hip_runtime.h>

#define NODES 1024
#define BATCH 64
#define FD    128
#define NEGV  (-9.0e15f)

typedef __attribute__((ext_vector_type(16))) _Float16 v16h;
typedef __attribute__((ext_vector_type(8)))  _Float16 h8;
typedef __attribute__((ext_vector_type(8)))  float    f8;

union V16H { v16h v; h8 h[2]; };

// ---------------------------------------------------------------------------
// K0: WT[k][o][f] = (f16) W[k][f][o]   (4*128*128 elements)
// ---------------------------------------------------------------------------
__global__ __launch_bounds__(256) void hg_wt_kernel(const float* __restrict__ W,
                                                    _Float16* __restrict__ WT) {
    int idx = blockIdx.x * 256 + threadIdx.x;        // 0 .. 65535
    int k   = idx >> 14;
    int rem = idx & 16383;
    int o   = rem >> 7;
    int f   = rem & 127;
    WT[idx] = (_Float16)W[((size_t)k * FD + f) * FD + o];
}

// ---------------------------------------------------------------------------
// K1: masked row-softmax -> A[k][i][j] (f16, row-major).  grid = (N, 4)
// ---------------------------------------------------------------------------
__global__ __launch_bounds__(256) void hg_softmax_kernel(
    const float* __restrict__ E0, const float* __restrict__ E1,
    const float* __restrict__ E2, const float* __restrict__ E3,
    const unsigned char* __restrict__ m0, const unsigned char* __restrict__ m1,
    const unsigned char* __restrict__ m2, const unsigned char* __restrict__ m3,
    _Float16* __restrict__ A) {
    int i = blockIdx.x;
    int k = blockIdx.y;
    const float* E = (k == 0) ? E0 : (k == 1) ? E1 : (k == 2) ? E2 : E3;
    const unsigned char* M = (k == 0) ? m0 : (k == 1) ? m1 : (k == 2) ? m2 : m3;
    int t = threadIdx.x;
    __shared__ float red[256];

    size_t rowoff = (size_t)i * NODES;
    float z[4];
#pragma unroll
    for (int u = 0; u < 4; ++u) {
        int j = u * 256 + t;
        z[u] = M[rowoff + j] ? E[rowoff + j] : NEGV;
    }
    float mx = fmaxf(fmaxf(z[0], z[1]), fmaxf(z[2], z[3]));
    red[t] = mx; __syncthreads();
    for (int s = 128; s > 0; s >>= 1) {
        if (t < s) red[t] = fmaxf(red[t], red[t + s]);
        __syncthreads();
    }
    mx = red[0]; __syncthreads();

    float e[4], lsum = 0.f;
#pragma unroll
    for (int u = 0; u < 4; ++u) { e[u] = __expf(z[u] - mx); lsum += e[u]; }
    red[t] = lsum; __syncthreads();
    for (int s = 128; s > 0; s >>= 1) {
        if (t < s) red[t] += red[t + s];
        __syncthreads();
    }
    float inv = 1.0f / red[0];

    _Float16* Arow = A + (size_t)k * NODES * NODES + rowoff;
#pragma unroll
    for (int u = 0; u < 4; ++u) {
        int j = u * 256 + t;
        Arow[j] = (_Float16)(e[u] * inv);
    }
}

// ---------------------------------------------------------------------------
// K2: hT[k][b][o][n] = (x @ W[k])^T in f16 via WMMA f16.
// grid = (65536/64, 4), block = 128 (4 waves, 16 rows each)
// ---------------------------------------------------------------------------
__global__ __launch_bounds__(128) void hg_gemm_xw_kernel(
    const float* __restrict__ x, const _Float16* __restrict__ WT,
    _Float16* __restrict__ hT) {
    int k    = blockIdx.y;
    int wave = threadIdx.x >> 5;
    int lane = threadIdx.x & 31;
    int c    = lane & 15;
    int hi   = lane >> 4;
    int r0   = blockIdx.x * 64 + wave * 16;   // flattened (b*N+n) row base
    int row  = r0 + c;                        // A-matrix row M = c

    // Preload 4 A fragments (K = 0..127 in steps of 32), fp32 -> f16
    V16H a[4];
    const float* xr = x + (size_t)row * FD;
#pragma unroll
    for (int ks = 0; ks < 4; ++ks) {
        const f8* p0 = (const f8*)(xr + ks * 32 + hi * 8);
        const f8* p1 = (const f8*)(xr + ks * 32 + hi * 8 + 16);
        a[ks].h[0] = __builtin_convertvector(*p0, h8);
        a[ks].h[1] = __builtin_convertvector(*p1, h8);
    }

    const _Float16* wtk = WT + (size_t)k * FD * FD;
    int b  = r0 >> 10;           // batch index (64-row block never crosses b)
    int n0 = r0 & (NODES - 1);
    _Float16* outBase = hT + ((size_t)k * BATCH + b) * FD * NODES;

#pragma unroll
    for (int ot = 0; ot < 8; ++ot) {
        int o = ot * 16 + c;                      // B-matrix column N = c
        const _Float16* wcol = wtk + (size_t)o * FD;
        f8 acc = {};
#pragma unroll
        for (int ks = 0; ks < 4; ++ks) {
            V16H bf;
            bf.h[0] = *(const h8*)(wcol + ks * 32 + hi * 8);
            bf.h[1] = *(const h8*)(wcol + ks * 32 + hi * 8 + 16);
            acc = __builtin_amdgcn_wmma_f32_16x16x32_f16(
                false, a[ks].v, false, bf.v, (short)0, acc, false, false);
        }
        // D lane holds 8 consecutive M rows for column o -> contiguous f16 store
        h8 st = __builtin_convertvector(acc, h8);
        *(h8*)(outBase + (size_t)o * NODES + n0 + hi * 8) = st;
    }
}

// ---------------------------------------------------------------------------
// K3: out[b][i][k*128+o] = sum_j A[k][i][j] * hT[k][b][o][j] + bias
// grid = (1024/64, 64, 4), block = 128.
// Each wave: 32 i-rows x 64 o-cols  -> per K-step: 2 A-frags + 4 B-frags
// feed 8 WMMAs (each B-frag reused by two back-to-back WMMAs).
// ---------------------------------------------------------------------------
__global__ __launch_bounds__(128) void hg_gemm_ah_kernel(
    const _Float16* __restrict__ A, const _Float16* __restrict__ hT,
    const float* __restrict__ bias, float* __restrict__ out) {
    int k    = blockIdx.z;
    int b    = blockIdx.y;
    int wave = threadIdx.x >> 5;
    int lane = threadIdx.x & 31;
    int c    = lane & 15;
    int hi   = lane >> 4;
    int iw   = wave & 1;                 // i-half of the 64x128 block tile
    int ow   = wave >> 1;                // o-half
    int i0   = blockIdx.x * 64 + iw * 32;
    int ob   = ow * 64;

    const _Float16* Ak = A + (size_t)k * NODES * NODES;
    const _Float16* ht = hT + ((size_t)k * BATCH + b) * FD * NODES;

    f8 acc[8] = {};                      // [it*4 + ot], it in {0,1}, ot in 0..3
    const _Float16* arow0 = Ak + (size_t)(i0 + c) * NODES;
    const _Float16* arow1 = arow0 + (size_t)16 * NODES;

    for (int js = 0; js < 32; ++js) {
        int j0 = js * 32;
        V16H a0, a1;
        a0.h[0] = *(const h8*)(arow0 + j0 + hi * 8);
        a0.h[1] = *(const h8*)(arow0 + j0 + hi * 8 + 16);
        a1.h[0] = *(const h8*)(arow1 + j0 + hi * 8);
        a1.h[1] = *(const h8*)(arow1 + j0 + hi * 8 + 16);
#pragma unroll
        for (int ot = 0; ot < 4; ++ot) {
            const _Float16* hcol =
                ht + (size_t)(ob + ot * 16 + c) * NODES + j0 + hi * 8;
            V16H bf;
            bf.h[0] = *(const h8*)hcol;
            bf.h[1] = *(const h8*)(hcol + 16);
            acc[ot] = __builtin_amdgcn_wmma_f32_16x16x32_f16(
                false, a0.v, false, bf.v, (short)0, acc[ot], false, false);
            acc[4 + ot] = __builtin_amdgcn_wmma_f32_16x16x32_f16(
                false, a1.v, false, bf.v, (short)0, acc[4 + ot], false, false);
        }
    }

    // Streaming output: non-temporal so hT/A stay resident in the 192MB L2.
#pragma unroll
    for (int it = 0; it < 2; ++it) {
        int irow = i0 + it * 16 + hi * 8;
#pragma unroll
        for (int ot = 0; ot < 4; ++ot) {
            int col  = k * FD + ob + ot * 16 + c;
            float bv = bias[col];
            f8 v = acc[it * 4 + ot];
#pragma unroll
            for (int r = 0; r < 8; ++r) {
                float* p = out + ((size_t)b * NODES + irow + r) * (4 * FD) + col;
                __builtin_nontemporal_store(v[r] + bv, p);
            }
        }
    }
}

// ---------------------------------------------------------------------------
extern "C" void kernel_launch(void* const* d_in, const int* in_sizes, int n_in,
                              void* d_out, int out_size, void* d_ws, size_t ws_size,
                              hipStream_t stream) {
    const float* x    = (const float*)d_in[0];
    const float* W    = (const float*)d_in[1];
    const float* E0   = (const float*)d_in[2];
    const float* E1   = (const float*)d_in[3];
    const float* E2   = (const float*)d_in[4];
    const float* E3   = (const float*)d_in[5];
    const float* bias = (const float*)d_in[6];
    const unsigned char* m0 = (const unsigned char*)d_in[7];   // jnp bool_ = 1 byte
    const unsigned char* m1 = (const unsigned char*)d_in[8];
    const unsigned char* m2 = (const unsigned char*)d_in[9];
    const unsigned char* m3 = (const unsigned char*)d_in[10];
    float* out = (float*)d_out;

    // Workspace layout (bytes):
    //   WT : 4*128*128*2        = 131072
    //   A  : 4*1024*1024*2      = 8388608
    //   hT : 4*64*128*1024*2    = 67108864
    char* ws = (char*)d_ws;
    _Float16* WT = (_Float16*)(ws);
    _Float16* Am = (_Float16*)(ws + 131072);
    _Float16* hT = (_Float16*)(ws + 131072 + 8388608);

    hg_wt_kernel<<<256, 256, 0, stream>>>(W, WT);

    dim3 gsm(NODES, 4);
    hg_softmax_kernel<<<gsm, 256, 0, stream>>>(E0, E1, E2, E3, m0, m1, m2, m3, Am);

    dim3 g1((BATCH * NODES) / 64, 4);
    hg_gemm_xw_kernel<<<g1, 128, 0, stream>>>(x, WT, hT);

    dim3 g2(NODES / 64, BATCH, 4);
    hg_gemm_ah_kernel<<<g2, 128, 0, stream>>>(Am, hT, bias, out);
}
